// BlockSparseMLP_82635170775195
// MI455X (gfx1250) — matrix operand order repaired and checked by reading, not compiled, not run on hardware
//
#include <hip/hip_runtime.h>
#include <cstdint>

// Problem constants (match reference)
#define TT 2048
#define DD 2048
#define FF 5632
#define EE 8
#define KK 2

typedef __attribute__((ext_vector_type(16))) __bf16        v16bf;
typedef __attribute__((ext_vector_type(2)))  __bf16        v2bf;
typedef __attribute__((ext_vector_type(2)))  float         v2f;
typedef __attribute__((ext_vector_type(8)))  float         v8f;
typedef __attribute__((ext_vector_type(8)))  unsigned int  v8u;

// ---------------- helpers ----------------

// fp32 pair -> packed bf16 dword; vector fptrunc selects v_cvt_pk_bf16_f32
__device__ __forceinline__ unsigned int pack2bf(float lo, float hi) {
  v2f f; f[0] = lo; f[1] = hi;
  v2bf b = __builtin_convertvector(f, v2bf);
  return __builtin_bit_cast(unsigned int, b);
}
__device__ __forceinline__ unsigned short f2bf(float f) {
  __bf16 b = (__bf16)f;
  return __builtin_bit_cast(unsigned short, b);
}

// A fragment 16x32 bf16 (contiguous k order, same as B) from LDS tile with row stride 64 halfs.
// lanes 0-15: row M=lane, K0..15 in v0..7;
// lanes 16-31: same rows, K16..31.
__device__ __forceinline__ v16bf load_a_frag64(const unsigned short* sA, int row, int ks, int lane) {
  const unsigned short* p = sA + row * 64 + ks + (lane >> 4) * 16;
  v8u u;
#pragma unroll
  for (int i = 0; i < 8; ++i) {
    u[i]     = *(const unsigned int*)(p + 2 * i);
  }
  return __builtin_bit_cast(v16bf, u);
}

// B fragment 32x16 bf16 from LDS [n][64 k] (k-contiguous) tile.
// lane = column (mod 16); lanes 0-15 K0..15 in v0..7, lanes 16-31 K16..31.
__device__ __forceinline__ v16bf load_b_frag64(const unsigned short* sBt, int col, int ks, int lane) {
  const unsigned short* p = sBt + col * 64 + ks + (lane >> 4) * 16;
  v8u u;
#pragma unroll
  for (int i = 0; i < 8; ++i) u[i] = *(const unsigned int*)(p + 2 * i);
  return __builtin_bit_cast(v16bf, u);
}

__device__ __forceinline__ v8f wmma_bf16(v16bf a, v16bf b, v8f c) {
  return __builtin_amdgcn_wmma_f32_16x16x32_bf16(false, a, false, b, (short)0, c, false, false);
}

// ---------------- kernel 1: zero out + counts ----------------

__global__ __launch_bounds__(256) void moe_zero(float* __restrict__ out, int* __restrict__ counts) {
  int i = blockIdx.x * 256 + threadIdx.x;
  if (i < TT * DD) out[i] = 0.0f;
  if (i < EE) counts[i] = 0;
}

// ---------------- kernel 2: router (softmax + top-2) ----------------

__global__ __launch_bounds__(256) void moe_router(const float* __restrict__ x,
                                                  const float* __restrict__ gate,
                                                  int* __restrict__ counts,
                                                  int* __restrict__ tokE,
                                                  float* __restrict__ tokW) {
  int t = blockIdx.x * 256 + threadIdx.x;
  if (t >= TT) return;
  const float* xr = x + (size_t)t * DD;
  float l[EE];
#pragma unroll
  for (int e = 0; e < EE; ++e) l[e] = 0.0f;
  for (int d = 0; d < DD; ++d) {
    float xv = xr[d];
    const float4 g0 = *(const float4*)(gate + (size_t)d * EE);
    const float4 g1 = *(const float4*)(gate + (size_t)d * EE + 4);
    l[0] += xv * g0.x; l[1] += xv * g0.y; l[2] += xv * g0.z; l[3] += xv * g0.w;
    l[4] += xv * g1.x; l[5] += xv * g1.y; l[6] += xv * g1.z; l[7] += xv * g1.w;
  }
  float m = l[0];
#pragma unroll
  for (int e = 1; e < EE; ++e) m = fmaxf(m, l[e]);
  float p[EE], S = 0.0f;
#pragma unroll
  for (int e = 0; e < EE; ++e) { p[e] = __expf(l[e] - m); S += p[e]; }
  float invS = 1.0f / S;
  int i0 = 0;
#pragma unroll
  for (int e = 1; e < EE; ++e) if (p[e] > p[i0]) i0 = e;
  int i1 = (i0 == 0) ? 1 : 0;
#pragma unroll
  for (int e = 0; e < EE; ++e) if (e != i0 && e != i1 && p[e] > p[i1]) i1 = e;
  float w0 = p[i0] * invS, w1 = p[i1] * invS;
  float rs = 1.0f / (w0 + w1 + 1e-20f);
  w0 *= rs; w1 *= rs;
  tokE[t * 2 + 0] = i0; tokE[t * 2 + 1] = i1;
  tokW[t * 2 + 0] = w0; tokW[t * 2 + 1] = w1;
  atomicAdd(&counts[i0], 1);
  atomicAdd(&counts[i1], 1);
}

// ---------------- kernel 3: scan counts -> offsets ----------------

__global__ void moe_scan(const int* __restrict__ counts, int* __restrict__ offsets,
                         int* __restrict__ cursor) {
  if (threadIdx.x == 0 && blockIdx.x == 0) {
    int acc = 0;
    for (int e = 0; e < EE; ++e) { offsets[e] = acc; cursor[e] = acc; acc += counts[e]; }
    offsets[EE] = acc;
  }
}

// ---------------- kernel 4: scatter tokens into expert slot lists ----------------

__global__ __launch_bounds__(256) void moe_scatter(const int* __restrict__ tokE,
                                                   const float* __restrict__ tokW,
                                                   int* __restrict__ cursor,
                                                   int* __restrict__ slotTok,
                                                   float* __restrict__ slotW) {
  int t = blockIdx.x * 256 + threadIdx.x;
  if (t >= TT) return;
#pragma unroll
  for (int k = 0; k < KK; ++k) {
    int e = tokE[t * 2 + k];
    int pos = atomicAdd(&cursor[e], 1);
    slotTok[pos] = t;
    slotW[pos] = tokW[t * 2 + k];
  }
}

// ---------------- kernel 5: grouped GEMM-1  h = silu(x@Wg) * (x@Wu) ----------------
// Block tile 64(M) x 128(N), K-step 64 over D (two WMMA K-sub-steps / barrier).
// 8 waves as 2(M) x 4(N): each wave owns 2 A-frags (32 rows) and 2 B-frags
// (32 cols) per matrix; B frags are reused across both A frags.
// Software pipeline: next K-slice global loads issued before the WMMA stage.

__global__ __launch_bounds__(256) void moe_gemm1(const float* __restrict__ x,
                                                 const float* __restrict__ Wg,
                                                 const float* __restrict__ Wu,
                                                 const int* __restrict__ offsets,
                                                 const int* __restrict__ slotTok,
                                                 unsigned short* __restrict__ hbuf) {
  const int e = blockIdx.z;
  const int rowBeg = offsets[e];
  const int rowEnd = offsets[e + 1];
  const int mBase = rowBeg + blockIdx.y * 64;
  if (mBase >= rowEnd) return;
  const int n0 = blockIdx.x * 128;
  const float* wg = Wg + (size_t)e * DD * FF;
  const float* wu = Wu + (size_t)e * DD * FF;

  __shared__ unsigned short sA[64 * 64];     // [row][k], 8 KB
  __shared__ unsigned short sBg[128 * 64];   // [n][k] k-contiguous, 16 KB
  __shared__ unsigned short sBu[128 * 64];   // 16 KB

  const int tid = threadIdx.x;
  const int lane = tid & 31;
  const int wave = tid >> 5;
  const int wmI = wave & 1;    // M half (32 rows)
  const int wnI = wave >> 1;   // N quarter (32 cols)

  v8f accG[2][2] = {}, accU[2][2] = {};

  // --- staging assignments (loop-invariant) ---
  // A: 64 rows x 64 k f32 = 1024 float4 chunks -> 4 per thread
  const int rA = tid >> 4;            // base row (0..15), +16 per chunk
  const int kq = (tid & 15) * 4;      // k offset
  const float* xrow[4];
#pragma unroll
  for (int c = 0; c < 4; ++c) {
    int s = mBase + rA + c * 16;
    xrow[c] = (s < rowEnd) ? (x + (size_t)slotTok[s] * DD) : nullptr;
  }
  // B: 32 k-pairs x 32 n-quads = 1024 chunks -> 4 per thread; each chunk loads
  // rows k=2p and k=2p+1 of columns n0+q*4..+3, packs k-pairs into dwords.
  int pP[4], qQ[4];
#pragma unroll
  for (int c = 0; c < 4; ++c) { int lin = tid + c * 256; pP[c] = lin >> 5; qQ[c] = lin & 31; }

  float4 ax[4];
  float4 gk0[4], gk1[4], uk0[4], uk1[4];

  auto load_stage = [&](int k0) {
#pragma unroll
    for (int c = 0; c < 4; ++c)
      ax[c] = xrow[c] ? *(const float4*)(xrow[c] + k0 + kq) : float4{0.f, 0.f, 0.f, 0.f};
#pragma unroll
    for (int c = 0; c < 4; ++c) {
      const float* bg = wg + (size_t)(k0 + 2 * pP[c]) * FF + n0 + qQ[c] * 4;
      gk0[c] = *(const float4*)bg;
      gk1[c] = *(const float4*)(bg + FF);
      const float* bu = wu + (size_t)(k0 + 2 * pP[c]) * FF + n0 + qQ[c] * 4;
      uk0[c] = *(const float4*)bu;
      uk1[c] = *(const float4*)(bu + FF);
    }
  };
  auto store_stage = [&]() {
#pragma unroll
    for (int c = 0; c < 4; ++c) {
      unsigned short* d = &sA[(rA + c * 16) * 64 + kq];
      *(unsigned int*)(d + 0) = pack2bf(ax[c].x, ax[c].y);
      *(unsigned int*)(d + 2) = pack2bf(ax[c].z, ax[c].w);
    }
#pragma unroll
    for (int c = 0; c < 4; ++c) {
      const float* g0 = (const float*)&gk0[c];
      const float* g1 = (const float*)&gk1[c];
      const float* u0 = (const float*)&uk0[c];
      const float* u1 = (const float*)&uk1[c];
#pragma unroll
      for (int j = 0; j < 4; ++j) {
        int n = qQ[c] * 4 + j;
        *(unsigned int*)(&sBg[n * 64 + 2 * pP[c]]) = pack2bf(g0[j], g1[j]);
        *(unsigned int*)(&sBu[n * 64 + 2 * pP[c]]) = pack2bf(u0[j], u1[j]);
      }
    }
  };

  load_stage(0);
  for (int k0 = 0; k0 < DD; k0 += 64) {
    store_stage();
    __syncthreads();
    if (k0 + 64 < DD) {
      load_stage(k0 + 64);   // overlap next global loads with WMMA below
      __builtin_prefetch(wg + (size_t)(k0 + 64) * FF + n0, 0, 0);
      __builtin_prefetch(wu + (size_t)(k0 + 64) * FF + n0, 0, 0);
    }
#pragma unroll
    for (int ks = 0; ks < 64; ks += 32) {
      v16bf a0 = load_a_frag64(sA, wmI * 32 + (lane & 15), ks, lane);
      v16bf a1 = load_a_frag64(sA, wmI * 32 + 16 + (lane & 15), ks, lane);
#pragma unroll
      for (int nf = 0; nf < 2; ++nf) {
        int col = wnI * 32 + nf * 16 + (lane & 15);
        v16bf bg = load_b_frag64(sBg, col, ks, lane);
        accG[0][nf] = wmma_bf16(a0, bg, accG[0][nf]);
        accG[1][nf] = wmma_bf16(a1, bg, accG[1][nf]);
        v16bf bu = load_b_frag64(sBu, col, ks, lane);
        accU[0][nf] = wmma_bf16(a0, bu, accU[0][nf]);
        accU[1][nf] = wmma_bf16(a1, bu, accU[1][nf]);
      }
    }
    __syncthreads();
  }

  // epilogue: silu(g)*u -> bf16 h
#pragma unroll
  for (int mf = 0; mf < 2; ++mf) {
#pragma unroll
    for (int nf = 0; nf < 2; ++nf) {
#pragma unroll
      for (int i = 0; i < 8; ++i) {
        int row = wmI * 32 + mf * 16 + i + (lane >> 4) * 8;
        int slot = mBase + row;
        if (slot < rowEnd) {
          int col = n0 + wnI * 32 + nf * 16 + (lane & 15);
          float g = accG[mf][nf][i], u = accU[mf][nf][i];
          float sig = 1.0f / (1.0f + __expf(-g));
          hbuf[(size_t)slot * FF + col] = f2bf(g * sig * u);
        }
      }
    }
  }
}

// ---------------- kernel 6: grouped GEMM-2  out[t] += w * (h @ Wd) ----------------

__global__ __launch_bounds__(256) void moe_gemm2(const unsigned short* __restrict__ hbuf,
                                                 const float* __restrict__ Wd,
                                                 const int* __restrict__ offsets,
                                                 const int* __restrict__ slotTok,
                                                 const float* __restrict__ slotW,
                                                 float* __restrict__ out) {
  const int e = blockIdx.z;
  const int rowBeg = offsets[e];
  const int rowEnd = offsets[e + 1];
  const int mBase = rowBeg + blockIdx.y * 64;
  if (mBase >= rowEnd) return;
  const int n0 = blockIdx.x * 128;
  const float* wd = Wd + (size_t)e * FF * DD;

  __shared__ unsigned short sA[64 * 64];   // 8 KB
  __shared__ unsigned short sB[128 * 64];  // 16 KB

  const int tid = threadIdx.x;
  const int lane = tid & 31;
  const int wave = tid >> 5;
  const int wmI = wave & 1;
  const int wnI = wave >> 1;

  v8f accD[2][2] = {};

  // A staging: 64 rows x 64 bf16 = 512 uint4 chunks -> 2 per thread
  const int rA = tid >> 3;            // base row (0..31), +32 for chunk 1
  const int k8 = (tid & 7) * 8;
  const unsigned short* hrow[2];
#pragma unroll
  for (int c = 0; c < 2; ++c) {
    int s = mBase + rA + c * 32;
    hrow[c] = (s < rowEnd) ? (hbuf + (size_t)s * FF) : nullptr;
  }
  // B staging: 32 k-pairs x 32 n-quads -> 4 chunks per thread
  int pP[4], qQ[4];
#pragma unroll
  for (int c = 0; c < 4; ++c) { int lin = tid + c * 256; pP[c] = lin >> 5; qQ[c] = lin & 31; }

  uint4 av[2];
  float4 dk0[4], dk1[4];

  auto load_stage = [&](int k0) {
#pragma unroll
    for (int c = 0; c < 2; ++c)
      av[c] = hrow[c] ? *(const uint4*)(hrow[c] + k0 + k8) : uint4{0u, 0u, 0u, 0u};
#pragma unroll
    for (int c = 0; c < 4; ++c) {
      const float* b = wd + (size_t)(k0 + 2 * pP[c]) * DD + n0 + qQ[c] * 4;
      dk0[c] = *(const float4*)b;
      dk1[c] = *(const float4*)(b + DD);
    }
  };
  auto store_stage = [&]() {
#pragma unroll
    for (int c = 0; c < 2; ++c)
      *(uint4*)(&sA[(rA + c * 32) * 64 + k8]) = av[c];
#pragma unroll
    for (int c = 0; c < 4; ++c) {
      const float* b0 = (const float*)&dk0[c];
      const float* b1 = (const float*)&dk1[c];
#pragma unroll
      for (int j = 0; j < 4; ++j) {
        int n = qQ[c] * 4 + j;
        *(unsigned int*)(&sB[n * 64 + 2 * pP[c]]) = pack2bf(b0[j], b1[j]);
      }
    }
  };

  load_stage(0);
  for (int k0 = 0; k0 < FF; k0 += 64) {
    store_stage();
    __syncthreads();
    if (k0 + 64 < FF) {
      load_stage(k0 + 64);
      __builtin_prefetch(wd + (size_t)(k0 + 64) * DD + n0, 0, 0);
    }
#pragma unroll
    for (int ks = 0; ks < 64; ks += 32) {
      v16bf a0 = load_a_frag64(sA, wmI * 32 + (lane & 15), ks, lane);
      v16bf a1 = load_a_frag64(sA, wmI * 32 + 16 + (lane & 15), ks, lane);
#pragma unroll
      for (int nf = 0; nf < 2; ++nf) {
        int col = wnI * 32 + nf * 16 + (lane & 15);
        v16bf b = load_b_frag64(sB, col, ks, lane);
        accD[0][nf] = wmma_bf16(a0, b, accD[0][nf]);
        accD[1][nf] = wmma_bf16(a1, b, accD[1][nf]);
      }
    }
    __syncthreads();
  }

#pragma unroll
  for (int mf = 0; mf < 2; ++mf) {
#pragma unroll
    for (int nf = 0; nf < 2; ++nf) {
#pragma unroll
      for (int i = 0; i < 8; ++i) {
        int row = wmI * 32 + mf * 16 + i + (lane >> 4) * 8;
        int slot = mBase + row;
        if (slot < rowEnd) {
          int t = slotTok[slot];
          float w = slotW[slot];
          int col = n0 + wnI * 32 + nf * 16 + (lane & 15);
          atomicAdd(&out[(size_t)t * DD + col], w * accD[mf][nf][i]);
        }
      }
    }
  }
}

// ---------------- host launcher ----------------

extern "C" void kernel_launch(void* const* d_in, const int* in_sizes, int n_in,
                              void* d_out, int out_size, void* d_ws, size_t ws_size,
                              hipStream_t stream) {
  const float* x    = (const float*)d_in[0];   // [T, D]
  const float* gate = (const float*)d_in[1];   // [D, E]
  const float* Wg   = (const float*)d_in[2];   // [E, D, F]
  const float* Wu   = (const float*)d_in[3];   // [E, D, F]
  const float* Wd   = (const float*)d_in[4];   // [E, F, D]
  float* out = (float*)d_out;                  // [T, D]

  char* ws = (char*)d_ws;
  size_t off = 0;
  auto take = [&](size_t bytes) { void* p = ws + off; off = (off + bytes + 255) & ~(size_t)255; return p; };
  int*            counts  = (int*)take(EE * sizeof(int));
  int*            cursor  = (int*)take(EE * sizeof(int));
  int*            offsets = (int*)take((EE + 1) * sizeof(int));
  int*            tokE    = (int*)take((size_t)TT * KK * sizeof(int));
  float*          tokW    = (float*)take((size_t)TT * KK * sizeof(float));
  int*            slotTok = (int*)take((size_t)TT * KK * sizeof(int));
  float*          slotW   = (float*)take((size_t)TT * KK * sizeof(float));
  unsigned short* hbuf    = (unsigned short*)take((size_t)TT * KK * FF * sizeof(unsigned short));
  (void)ws_size; (void)in_sizes; (void)n_in; (void)out_size;

  moe_zero<<<(TT * DD + 255) / 256, 256, 0, stream>>>(out, counts);
  moe_router<<<TT / 256, 256, 0, stream>>>(x, gate, counts, tokE, tokW);
  moe_scan<<<1, 32, 0, stream>>>(counts, offsets, cursor);
  moe_scatter<<<TT / 256, 256, 0, stream>>>(tokE, tokW, cursor, slotTok, slotW);

  dim3 g1(FF / 128, (TT + 63) / 64, EE);   // (44, 32, 8)
  moe_gemm1<<<g1, 256, 0, stream>>>(x, Wg, Wu, offsets, slotTok, hbuf);

  dim3 g2(DD / 128, (TT + 63) / 64, EE);   // (16, 32, 8)
  moe_gemm2<<<g2, 256, 0, stream>>>(hbuf, Wd, offsets, slotTok, slotW, out);
}
